// LSTM1_ts_73469710565550
// MI455X (gfx1250) — compile-verified, matrix operand order checked by reading, not run on hardware
//
#include <hip/hip_runtime.h>
#include <cstdint>
#include <cstddef>

// ---------------------------------------------------------------------------
// LSTM (B=512, T=512, F=128, H=512, NT=1) fused persistent kernel for gfx1250
// ---------------------------------------------------------------------------
#define BATCH 512
#define SEQ   512
#define FEAT  128
#define HID   512
#define NBLK  128          // persistent workgroups: 16 row-blocks x 8 col-blocks
#define LDS_ROWSTRIDE 520  // 512 bf16 + 8 pad (1040 B, 16B aligned)
#define LDS_BYTES (4 * 64 * LDS_ROWSTRIDE * 2)  // 266,240 B

typedef __attribute__((ext_vector_type(16))) __bf16 v16bf;
typedef __attribute__((ext_vector_type(8)))  float  v8f;
typedef __attribute__((ext_vector_type(4)))  int    v4i;
typedef __attribute__((ext_vector_type(4)))  unsigned int v4u;
typedef __attribute__((address_space(1))) v4i gv4i;            // global 16B vec
typedef __attribute__((address_space(3))) v4i lv4i;            // LDS 16B vec
typedef const __attribute__((address_space(1))) v4u cgv4u;     // global 16B load
typedef __attribute__((address_space(1))) uint16_t gu16;       // global u16

union FragU { v4u q[2]; v16bf v; };

#if defined(__has_builtin)
#  if __has_builtin(__builtin_amdgcn_global_load_async_to_lds_b128) && \
      __has_builtin(__builtin_amdgcn_s_wait_asynccnt)
#    define USE_ASYNC_LDS 1
#  endif
#endif

__device__ __forceinline__ uint16_t f2bf(float f) {
  union { float f; uint32_t u; } v; v.f = f;
  uint32_t r = v.u + 0x7FFFu + ((v.u >> 16) & 1u);   // round-to-nearest-even
  return (uint16_t)(r >> 16);
}

__device__ __forceinline__ float sigmoidf_(float x) {
  return 1.0f / (1.0f + __expf(-x));
}

// 16x32 bf16 WMMA fragment loads. rowp = this lane's row start (element units).
// Lane L in [0,15]: K chunks {k0..k0+7},{k0+16..k0+23};
// lane L+16: {k0+8..k0+15},{k0+24..k0+31}.  half8 = (lane>>4)*8.
// Global-memory variant: force address_space(1) so backend emits global_load_b128
// (NOT flat_load, which would also tie up DScnt and the LDS path).
__device__ __forceinline__ v16bf load_frag_g(const uint16_t* rowp, int k0, int half8) {
  FragU u;
  u.q[0] = *(cgv4u*)(rowp + k0 + half8);
  u.q[1] = *(cgv4u*)(rowp + k0 + half8 + 16);
  return u.v;
}
// LDS variant (generic pointer into __shared__ lowers to ds_load_b128).
__device__ __forceinline__ v16bf load_frag_l(const uint16_t* rowp, int k0, int half8) {
  FragU u;
  u.q[0] = *(const v4u*)(rowp + k0 + half8);
  u.q[1] = *(const v4u*)(rowp + k0 + half8 + 16);
  return u.v;
}

// -------------------------- helper kernels ---------------------------------
__global__ void k_f32_to_bf16_v4(const float* __restrict__ in,
                                 uint16_t* __restrict__ out, int n4) {
  int i = blockIdx.x * blockDim.x + threadIdx.x;
  if (i >= n4) return;
  float4 f = ((const float4*)in)[i];
  union { uint16_t s[4]; uint2 u; } r;
  r.s[0] = f2bf(f.x); r.s[1] = f2bf(f.y); r.s[2] = f2bf(f.z); r.s[3] = f2bf(f.w);
  ((uint2*)out)[i] = r.u;
}

__global__ void k_init(uint16_t* __restrict__ h0, uint16_t* __restrict__ h1,
                       float* __restrict__ out, const float* __restrict__ b_out,
                       unsigned* __restrict__ bar) {
  int i = blockIdx.x * blockDim.x + threadIdx.x;
  if (i < BATCH * HID) { h0[i] = 0; h1[i] = 0; }
  if (i < BATCH * SEQ) out[i] = b_out[0];
  if (i < 2) bar[i] = 0;
}

// --------------------------- grid barrier ----------------------------------
__device__ void grid_barrier(unsigned* cnt, unsigned* gen, unsigned nblocks) {
  __syncthreads();
  if (threadIdx.x == 0) {
    __threadfence();
    unsigned g = __hip_atomic_load(gen, __ATOMIC_ACQUIRE, __HIP_MEMORY_SCOPE_AGENT);
    if (atomicAdd(cnt, 1u) == nblocks - 1u) {
      __hip_atomic_store(cnt, 0u, __ATOMIC_RELAXED, __HIP_MEMORY_SCOPE_AGENT);
      __hip_atomic_fetch_add(gen, 1u, __ATOMIC_RELEASE, __HIP_MEMORY_SCOPE_AGENT);
    } else {
      while (__hip_atomic_load(gen, __ATOMIC_ACQUIRE, __HIP_MEMORY_SCOPE_AGENT) == g)
        __builtin_amdgcn_s_sleep(2);
    }
  }
  __syncthreads();
}

// ------------------------- persistent LSTM kernel --------------------------
// grid = 128 blocks x 256 threads (8 waves of 32).
// blockIdx.x -> mb (row block of 32 batches) = bid>>3 ; nb (64 hidden cols) = bid&7
// wave w: wm=w&1 (row sub-tile), wn=w>>1 (col sub-tile 0..3): one 16x16 tile/gate.
__global__ void __launch_bounds__(256)
lstm_persistent(const uint16_t* __restrict__ Xbf,     // [B][T][F] bf16
                const uint16_t* __restrict__ Wih,     // [4H][F]   bf16
                const uint16_t* __restrict__ Whh,     // [4H][H]   bf16
                const float* __restrict__ b_ih, const float* __restrict__ b_hh,
                const float* __restrict__ W_out,      // [1][H] f32
                uint16_t* __restrict__ h0, uint16_t* __restrict__ h1,
                float* __restrict__ out,              // [B][T] f32 (pre-set to b_out)
                unsigned* __restrict__ bar_cnt, unsigned* __restrict__ bar_gen) {
  extern __shared__ __align__(16) uint16_t smem[];   // W_hh slice: (4*64) x 520 bf16

  const int tid  = threadIdx.x;
  const int lane = tid & 31;
  const int w    = tid >> 5;
  const int bid  = blockIdx.x;
  const int mb   = bid >> 3;
  const int nb   = bid & 7;
  const int wm   = w & 1;
  const int wn   = w >> 1;
  const int m0   = mb * 32 + wm * 16;        // batch row base of this wave's tile
  const int hc0  = nb * 64 + wn * 16;        // hidden col base of this wave's tile
  const int l15  = lane & 15;
  const int half8 = (lane >> 4) << 3;        // 0 or 8
  const int col  = hc0 + l15;                // this lane's hidden column
  const int rbase = m0 + (half8);            // C/D fragment row base for this lane

  // ---- Stage this WG's W_hh slice (4 gates x 64 cols x 512 K bf16) to LDS ----
  {
    const int rr = tid;                      // 256 rows, one per thread
    const int g  = rr >> 6, cl = rr & 63;
    const uint16_t* src = Whh + ((size_t)(g * HID + nb * 64 + cl)) * HID;
    uint16_t* dst = smem + (size_t)rr * LDS_ROWSTRIDE;
#if defined(USE_ASYNC_LDS)
    for (int c = 0; c < 64; ++c) {
      __builtin_amdgcn_global_load_async_to_lds_b128(
          (gv4i*)(src + c * 8), (lv4i*)(dst + c * 8), 0, 0);
    }
    __builtin_amdgcn_s_wait_asynccnt(0);
#else
    for (int c = 0; c < 64; ++c)
      *(v4u*)(dst + c * 8) = *(cgv4u*)(src + c * 8);
#endif
  }
  __syncthreads();

  // ---- per-lane invariants ----
  const uint16_t* bihrow[4];
  const uint16_t* ldsrow[4];
  float bias[4];
#pragma unroll
  for (int g = 0; g < 4; ++g) {
    bihrow[g] = Wih + ((size_t)(g * HID + col)) * FEAT;
    ldsrow[g] = smem + (size_t)(g * 64 + wn * 16 + l15) * LDS_ROWSTRIDE;
    bias[g]   = b_ih[g * HID + col] + b_hh[g * HID + col];
  }
  const float wout = W_out[col];
  const size_t hoff = (size_t)(m0 + l15) * HID;     // A-fragment row into h buffer

  float cc[8];
#pragma unroll
  for (int r = 0; r < 8; ++r) cc[r] = 0.0f;

  uint16_t* hb[2] = { h0, h1 };

#pragma unroll 1
  for (int t = 0; t < SEQ; ++t) {
    const int p = t & 1;                      // read hb[p], write hb[p^1]
    v8f acc[4] = {};

    // ---- gates += X_t . W_ih^T  (K = 128) ----
    const uint16_t* xrow = Xbf + ((size_t)(m0 + l15) * SEQ + t) * FEAT;
#pragma unroll
    for (int k0 = 0; k0 < FEAT; k0 += 32) {
      v16bf a = load_frag_g(xrow, k0, half8);
#pragma unroll
      for (int g = 0; g < 4; ++g) {
        v16bf b = load_frag_g(bihrow[g], k0, half8);
        acc[g] = __builtin_amdgcn_wmma_f32_16x16x32_bf16(
            false, a, false, b, (short)0, acc[g], false, false);
      }
    }

    // ---- gates += h . W_hh^T  (K = 512, B-fragments from LDS) ----
    const uint16_t* hrow = hb[p] + hoff;
#pragma unroll 4
    for (int k0 = 0; k0 < HID; k0 += 32) {
      v16bf a = load_frag_g(hrow, k0, half8);   // global_load_b128, not flat
#pragma unroll
      for (int g = 0; g < 4; ++g) {
        v16bf b = load_frag_l(ldsrow[g], k0, half8);
        acc[g] = __builtin_amdgcn_wmma_f32_16x16x32_bf16(
            false, a, false, b, (short)0, acc[g], false, false);
      }
    }

    // ---- elementwise LSTM cell update (c lives in registers) ----
    float part[8];
    gu16* hnext = (gu16*)hb[p ^ 1];           // force global_store_b16
#pragma unroll
    for (int r = 0; r < 8; ++r) {
      float iv = sigmoidf_(acc[0][r] + bias[0]);
      float fv = sigmoidf_(acc[1][r] + bias[1]);
      float gv = tanhf    (acc[2][r] + bias[2]);
      float ov = sigmoidf_(acc[3][r] + bias[3]);
      float c  = fv * cc[r] + iv * gv;
      cc[r] = c;
      float hv = ov * tanhf(c);
      hnext[(size_t)(rbase + r) * HID + col] = f2bf(hv);
      part[r] = hv * wout;                    // partial of h . W_out^T
    }

    // reduce partials over the 16 columns of this tile (within each half-wave)
#pragma unroll
    for (int mask = 1; mask <= 8; mask <<= 1) {
#pragma unroll
      for (int r = 0; r < 8; ++r)
        part[r] += __shfl_xor(part[r], mask, 32);
    }
    if (l15 == 0) {
#pragma unroll
      for (int r = 0; r < 8; ++r)
        atomicAdd(&out[(size_t)(rbase + r) * SEQ + t], part[r]);
    }

    // all WGs must see the new h before the next timestep
    grid_barrier(bar_cnt, bar_gen, NBLK);
  }
}

// ------------------------------ launcher -----------------------------------
extern "C" void kernel_launch(void* const* d_in, const int* in_sizes, int n_in,
                              void* d_out, int out_size, void* d_ws, size_t ws_size,
                              hipStream_t stream) {
  (void)in_sizes; (void)n_in; (void)out_size; (void)ws_size;
  const float* X     = (const float*)d_in[0];
  const float* W_ih  = (const float*)d_in[1];
  const float* W_hh  = (const float*)d_in[2];
  const float* b_ih  = (const float*)d_in[3];
  const float* b_hh  = (const float*)d_in[4];
  const float* W_out = (const float*)d_in[5];
  const float* b_out = (const float*)d_in[6];
  float* out = (float*)d_out;

  uint8_t* ws = (uint8_t*)d_ws;
  // workspace layout (bytes)
  const size_t nX   = (size_t)BATCH * SEQ * FEAT;   // 33,554,432 elems
  const size_t nWih = (size_t)4 * HID * FEAT;       //    262,144
  const size_t nWhh = (size_t)4 * HID * HID;        //  1,048,576
  const size_t nH   = (size_t)BATCH * HID;          //    262,144
  uint16_t* Xbf    = (uint16_t*)ws;
  uint16_t* Wih_bf = (uint16_t*)(ws + nX * 2);
  uint16_t* Whh_bf = (uint16_t*)(ws + nX * 2 + nWih * 2);
  uint16_t* h0     = (uint16_t*)(ws + nX * 2 + nWih * 2 + nWhh * 2);
  uint16_t* h1     = (uint16_t*)(ws + nX * 2 + nWih * 2 + nWhh * 2 + nH * 2);
  unsigned* bar    = (unsigned*)(ws + nX * 2 + nWih * 2 + nWhh * 2 + nH * 4);

  // f32 -> bf16 conversions
  k_f32_to_bf16_v4<<<(unsigned)((nX / 4 + 255) / 256), 256, 0, stream>>>(X, Xbf, (int)(nX / 4));
  k_f32_to_bf16_v4<<<(unsigned)((nWih / 4 + 255) / 256), 256, 0, stream>>>(W_ih, Wih_bf, (int)(nWih / 4));
  k_f32_to_bf16_v4<<<(unsigned)((nWhh / 4 + 255) / 256), 256, 0, stream>>>(W_hh, Whh_bf, (int)(nWhh / 4));

  // zero h state, preset out = b_out, reset barrier
  k_init<<<(BATCH * SEQ + 255) / 256, 256, 0, stream>>>(h0, h1, out, b_out, bar);

  // persistent fused LSTM
  lstm_persistent<<<NBLK, 256, LDS_BYTES, stream>>>(
      Xbf, Wih_bf, Whh_bf, b_ih, b_hh, W_out, h0, h1, out, bar, bar + 1);
}
